// HybridTransformerClassifier_65481071405771
// MI455X (gfx1250) — matrix operand order repaired and checked by reading, not compiled, not run on hardware
//
#include <hip/hip_runtime.h>
#include <hip/hip_bf16.h>

typedef __attribute__((ext_vector_type(16))) _Float16 v16h;
typedef __attribute__((ext_vector_type(8)))  _Float16 v8h;
typedef __attribute__((ext_vector_type(8)))  float    v8f;

#define B_  8
#define S_  2048
#define E_  256
#define L_  4
#define FF_ 1024
#define NQ_ 8
#define NC_ 10
#define NH_ 32
#define EPS_ 1e-5f

#define WMMA(a, b, c) __builtin_amdgcn_wmma_f32_16x16x32_f16(false, (a), false, (b), (short)0, (c), false, false)

__device__ __forceinline__ v16h mk16(v8h lo, v8h hi) {
  v16h r;
#pragma unroll
  for (int i = 0; i < 8; i++) { r[i] = lo[i]; r[i + 8] = hi[i]; }
  return r;
}

// ---------------- embedding + positional encoding ----------------
__global__ void k_embed(const int* __restrict__ tok, const float* __restrict__ emb,
                        float* __restrict__ x) {
  int bs = blockIdx.x;             // b*S + s
  int e  = threadIdx.x;            // 0..255
  int s  = bs % S_;
  int t  = tok[bs];
  // div = exp(-(2i)*ln(10000)/E), i = e/2 ; ln(10000)/256 = 0.03597828f
  float ang = (float)s * __expf(-(float)(e & ~1) * 0.03597828f);
  float pe  = (e & 1) ? cosf(ang) : sinf(ang);
  x[(size_t)bs * E_ + e] = emb[(size_t)t * E_ + e] + pe;
}

// ---------------- quantum attention feature map (q) ----------------
// q per 8-wire group: c_j = cos(x_j + phi_j); q0 = prod_{j>=1} c_j; q_i = cumprod_{j<=i} c_j
__global__ void k_q(const float* __restrict__ x, const float* __restrict__ phi, int l,
                    _Float16* __restrict__ qh) {
  int gid = blockIdx.x * 256 + threadIdx.x;     // over B*S*NH
  int bs = gid / NH_, hg = gid % NH_;
  const float* xr = x + (size_t)bs * E_ + hg * 8;
  float c[8], cum[8], q[8];
#pragma unroll
  for (int j = 0; j < 8; j++) c[j] = cosf(xr[j] + phi[l * NQ_ + j]);
  cum[0] = c[0];
#pragma unroll
  for (int j = 1; j < 8; j++) cum[j] = cum[j - 1] * c[j];
  float prod1 = 1.f;
#pragma unroll
  for (int j = 1; j < 8; j++) prod1 *= c[j];
  q[0] = prod1;
#pragma unroll
  for (int j = 1; j < 8; j++) q[j] = cum[j];
  v8h hv;
#pragma unroll
  for (int j = 0; j < 8; j++) hv[j] = (_Float16)q[j];
  *(v8h*)(qh + (size_t)bs * E_ + hg * 8) = hv;           // [b][s][e] contiguous
}

// ---------------- LDS-tiled transpose qh[b][s][e] -> qhT[b][e][s] ----------------
// 64x64 f16 tile per block; coalesced 32B loads and stores on both sides.
__global__ void k_qt(const _Float16* __restrict__ qh, _Float16* __restrict__ qhT) {
  __shared__ _Float16 tile[64][72];             // pad 8 halves to spread banks
  int b  = blockIdx.z;
  int s0 = blockIdx.x * 64;
  int e0 = blockIdx.y * 64;
  int tid = threadIdx.x;
  int r = tid >> 2;                             // 0..63
  int c = (tid & 3) * 16;                       // 0,16,32,48
  const _Float16* src = qh + ((size_t)b * S_ + s0 + r) * E_ + e0 + c;
  v8h a0 = *(const v8h*)(src);
  v8h a1 = *(const v8h*)(src + 8);
#pragma unroll
  for (int j = 0; j < 8; j++) { tile[r][c + j] = a0[j]; tile[r][c + 8 + j] = a1[j]; }
  __syncthreads();
  v8h w0, w1;
#pragma unroll
  for (int j = 0; j < 8; j++) { w0[j] = tile[c + j][r]; w1[j] = tile[c + 8 + j][r]; }
  _Float16* dst = qhT + ((size_t)b * E_ + e0 + r) * S_ + s0 + c;
  *(v8h*)(dst)     = w0;
  *(v8h*)(dst + 8) = w1;
}

// ---------------- flash attention on WMMA (transposed tiles) ----------------
// Workgroup: one (b, 16-query tile). 8 waves split 64 key-blocks of 32 keys.
// Sᵀ = K·Qᵀ (M=keys, N=queries); Oᵀ = Vᵀ·Pᵀ (M=embed, N=queries).
__launch_bounds__(256)
__global__ void k_attn(const _Float16* __restrict__ qh, const _Float16* __restrict__ qhT,
                       float* __restrict__ aout) {
  const int b   = blockIdx.y;
  const int q0  = blockIdx.x * 16;
  const int tid = threadIdx.x;
  const int lane = tid & 31, wid = tid >> 5;
  const int lm  = lane & 15;
  const bool hi = lane >= 16;
  const float scale = 0.35355339059327373f;   // 1/sqrt(8)

  __shared__ float Osum[E_ * 16];             // [e][q] 16KB
  __shared__ float mS[8][16], lS[8][16];
  for (int i = tid; i < E_ * 16; i += 256) Osum[i] = 0.f;
  __syncthreads();

  // Resident Qᵀ B-fragments: lane=query column, halves=contiguous e
  const _Float16* qrow = qh + ((size_t)b * S_ + q0 + lm) * E_;
  v16h qb[8];
#pragma unroll
  for (int ks = 0; ks < 8; ks++)
    qb[ks] = *(const v16h*)(qrow + 32 * ks + (hi ? 16 : 0));

  v8f o[16];
#pragma unroll
  for (int t = 0; t < 16; t++) o[t] = (v8f){};
  float m = -3.0e38f, lsum = 0.f;
  const int e0 = hi ? 16 : 0;                 // A-fragment K-chunk select
  const _Float16* vbase0 = qhT + ((size_t)b * E_ + lm) * S_;

  for (int kb = wid; kb < 64; kb += 8) {
    const int kbase = kb * 32;
    const _Float16* k0row = qh + ((size_t)b * S_ + kbase + lm) * E_;
    const _Float16* k1row = k0row + (size_t)16 * E_;
    // pull next block's K rows / V columns toward the WGP while VALU runs
    if (kb < 56) {
      __builtin_prefetch(k0row + (size_t)256 * E_, 0, 3);
      __builtin_prefetch(k1row + (size_t)256 * E_, 0, 3);
      __builtin_prefetch(vbase0 + kbase + 256, 0, 3);
    }
    v8f s0 = (v8f){}, s1 = (v8f){};
#pragma unroll
    for (int ks = 0; ks < 8; ks++) {
      v16h a0 = mk16(*(const v8h*)(k0row + 32 * ks + e0),
                     *(const v8h*)(k0row + 32 * ks + e0 + 8));
      v16h a1 = mk16(*(const v8h*)(k1row + 32 * ks + e0),
                     *(const v8h*)(k1row + 32 * ks + e0 + 8));
      s0 = WMMA(a0, qb[ks], s0);
      s1 = WMMA(a1, qb[ks], s1);
    }
    // online softmax over keys (M dim): per-lane + one half-wave swap
    float bm = -3.0e38f;
#pragma unroll
    for (int r = 0; r < 8; r++) bm = fmaxf(bm, fmaxf(s0[r], s1[r]));
    bm *= scale;
    bm = fmaxf(bm, __shfl_xor(bm, 16));
    float mn = fmaxf(m, bm);
    float alpha = __expf(m - mn);
    m = mn;
    float p0[8], p1[8], ps = 0.f;
#pragma unroll
    for (int r = 0; r < 8; r++) {
      p0[r] = __expf(s0[r] * scale - mn);
      p1[r] = __expf(s1[r] * scale - mn);
      ps += p0[r] + p1[r];
    }
    lsum = lsum * alpha + ps;
#pragma unroll
    for (int t = 0; t < 16; t++)
#pragma unroll
      for (int r = 0; r < 8; r++) o[t][r] *= alpha;
    // Pᵀ B-fragment: swap halves between lane pairs (L, L^16)
    v16h bp;
#pragma unroll
    for (int h = 0; h < 8; h++) {
      float y0 = __shfl_xor(p0[h], 16);
      float y1 = __shfl_xor(p1[h], 16);
      bp[h]     = (_Float16)(hi ? y1 : p0[h]);
      bp[h + 8] = (_Float16)(hi ? p1[h] : y0);
    }
    // Oᵀ += Vᵀ·Pᵀ over 16 embed tiles (2-stage pipelined)
    const _Float16* vb = vbase0 + kbase + e0;
    v8h vlo = *(const v8h*)(vb);
    v8h vhi = *(const v8h*)(vb + 8);
#pragma unroll
    for (int t = 0; t < 16; t++) {
      v16h av = mk16(vlo, vhi);
      if (t < 15) {
        vlo = *(const v8h*)(vb + (size_t)(t + 1) * 16 * S_);
        vhi = *(const v8h*)(vb + (size_t)(t + 1) * 16 * S_ + 8);
      }
      o[t] = WMMA(av, bp, o[t]);
    }
  }

  // cross-wave combine
  float lt = lsum + __shfl_xor(lsum, 16);
  if (!hi) { mS[wid][lm] = m; lS[wid][lm] = lt; }
  __syncthreads();
  float M = -3.0e38f;
#pragma unroll
  for (int w = 0; w < 8; w++) M = fmaxf(M, mS[w][lm]);
  float D = 0.f;
#pragma unroll
  for (int w = 0; w < 8; w++) D += lS[w][lm] * __expf(mS[w][lm] - M);
  float sc = __expf(m - M) / D;
#pragma unroll
  for (int t = 0; t < 16; t++)
#pragma unroll
    for (int r = 0; r < 8; r++)
      atomicAdd(&Osum[(t * 16 + r + (hi ? 8 : 0)) * 16 + lm], o[t][r] * sc);
  __syncthreads();
  {
    int q = tid >> 4, c0 = tid & 15;
    float* dst = aout + ((size_t)b * S_ + q0 + q) * E_;
#pragma unroll
    for (int j = 0; j < 16; j++) {
      int e = c0 + 16 * j;
      dst[e] = Osum[e * 16 + q];
    }
  }
}

// ---------------- residual + layernorm (in-place capable) ----------------
__global__ void k_ln(const float* __restrict__ x, const float* __restrict__ d,
                     const float* __restrict__ g, const float* __restrict__ be,
                     float* __restrict__ out) {
  int row = blockIdx.x, t = threadIdx.x;
  float v = x[(size_t)row * E_ + t] + d[(size_t)row * E_ + t];
  __shared__ float rs[256], rq[256];
  rs[t] = v; rq[t] = v * v;
  __syncthreads();
  for (int s = 128; s > 0; s >>= 1) {
    if (t < s) { rs[t] += rs[t + s]; rq[t] += rq[t + s]; }
    __syncthreads();
  }
  float mean = rs[0] * (1.f / E_);
  float var  = rq[0] * (1.f / E_) - mean * mean;
  float r = rsqrtf(var + EPS_);
  out[(size_t)row * E_ + t] = (v - mean) * r * g[t] + be[t];
}

// ---------------- W2 -> W2ᵀ (f16) prep ----------------
__global__ void k_w2t(const float* __restrict__ W2, _Float16* __restrict__ W2T) {
  size_t i = (size_t)blockIdx.x * 256 + threadIdx.x;  // target [l][e][k]
  int l = (int)(i / ((size_t)E_ * FF_));
  int rem = (int)(i % ((size_t)E_ * FF_));
  int e = rem / FF_, k = rem % FF_;
  W2T[i] = (_Float16)W2[((size_t)l * FF_ + k) * E_ + e];
}

// ---------------- quantum FFN: qm -> h (VALU) -> h·W2 (WMMA, transposed) ----------------
__launch_bounds__(256)
__global__ void k_ffn(const float* __restrict__ x, const float* __restrict__ theta,
                      const float* __restrict__ W1, const float* __restrict__ b1,
                      const _Float16* __restrict__ W2T, const float* __restrict__ b2,
                      int l, float* __restrict__ dout) {
  int r0 = blockIdx.x * 16;                  // row tile in [0, B*S)
  int tid = threadIdx.x;
  int lane = tid & 31, wid = tid >> 5, lm = lane & 15;
  bool hi = lane >= 16;
  __shared__ _Float16 hs[16][FF_];           // 32KB hidden tile (f16)
  __shared__ float fs[16][E_];               // 16KB output staging
  __shared__ float qs[16][8];

  if (tid < 128) {
    int r = tid >> 3, i = tid & 7;
    qs[r][i] = cosf(x[(size_t)(r0 + r) * E_ + i]) * cosf(theta[l * NQ_ + i]);
  }
  __syncthreads();

  const float* W1l = W1 + (size_t)l * NQ_ * FF_;
  const float* b1l = b1 + (size_t)l * FF_;
  for (int idx = tid; idx < 16 * FF_; idx += 256) {
    int r = idx >> 10, j = idx & (FF_ - 1);
    float acc = b1l[j];
#pragma unroll
    for (int i = 0; i < 8; i++) acc += qs[r][i] * W1l[i * FF_ + j];
    hs[r][j] = (_Float16)fmaxf(acc, 0.f);
  }
  __syncthreads();

  // fᵀ = W2ᵀ · hᵀ : wave owns embed tiles [2w, 2w+1], K = 1024 in 32 steps
  v8f c0 = (v8f){}, c1 = (v8f){};
  const _Float16* Wl = W2T + (size_t)l * E_ * FF_;
  int e0 = wid * 32;
  const _Float16* ar0 = Wl + (size_t)(e0 + lm) * FF_;
  const _Float16* ar1 = Wl + (size_t)(e0 + 16 + lm) * FF_;
  const int ka = hi ? 16 : 0;
  for (int ks = 0; ks < 32; ks++) {
    v16h bfr = *(const v16h*)&hs[lm][32 * ks + (hi ? 16 : 0)];
    v16h a0 = mk16(*(const v8h*)(ar0 + 32 * ks + ka),
                   *(const v8h*)(ar0 + 32 * ks + ka + 8));
    v16h a1 = mk16(*(const v8h*)(ar1 + 32 * ks + ka),
                   *(const v8h*)(ar1 + 32 * ks + ka + 8));
    c0 = WMMA(a0, bfr, c0);
    c1 = WMMA(a1, bfr, c1);
  }
#pragma unroll
  for (int r = 0; r < 8; r++) {
    fs[lm][e0 + r + (hi ? 8 : 0)]      = c0[r];
    fs[lm][e0 + 16 + r + (hi ? 8 : 0)] = c1[r];
  }
  __syncthreads();
  const float* b2l = b2 + (size_t)l * E_;
  {
    int r = tid >> 4, cb = tid & 15;
    float* dst = dout + (size_t)(r0 + r) * E_;
#pragma unroll
    for (int j = 0; j < 16; j++) {
      int e = cb + 16 * j;
      dst[e] = fs[r][e] + b2l[e];
    }
  }
}

// ---------------- mean over S + classifier head ----------------
__global__ void k_cls(const float* __restrict__ x, const float* __restrict__ Wc,
                      const float* __restrict__ bc, float* __restrict__ out) {
  int b = blockIdx.x, t = threadIdx.x;
  float acc = 0.f;
  for (int s = 0; s < S_; s++) acc += x[((size_t)b * S_ + s) * E_ + t];
  __shared__ float ms[E_];
  ms[t] = acc * (1.f / S_);
  __syncthreads();
  if (t < NC_) {
    float o = bc[t];
    for (int e = 0; e < E_; e++) o += ms[e] * Wc[e * NC_ + t];
    out[b * NC_ + t] = o;
  }
}

extern "C" void kernel_launch(void* const* d_in, const int* in_sizes, int n_in,
                              void* d_out, int out_size, void* d_ws, size_t ws_size,
                              hipStream_t stream) {
  (void)in_sizes; (void)n_in; (void)out_size; (void)ws_size;
  const int*   tokens = (const int*)d_in[0];
  const float* emb   = (const float*)d_in[1];
  const float* phi   = (const float*)d_in[2];
  const float* theta = (const float*)d_in[3];
  const float* W1    = (const float*)d_in[4];
  const float* b1    = (const float*)d_in[5];
  const float* W2    = (const float*)d_in[6];
  const float* b2    = (const float*)d_in[7];
  const float* g1    = (const float*)d_in[8];
  const float* be1   = (const float*)d_in[9];
  const float* g2    = (const float*)d_in[10];
  const float* be2   = (const float*)d_in[11];
  const float* Wc    = (const float*)d_in[12];
  const float* bc    = (const float*)d_in[13];
  float* out = (float*)d_out;

  char* ws = (char*)d_ws;
  size_t off = 0;
  auto alloc = [&](size_t bytes) -> void* {
    void* p = ws + off;
    off = (off + bytes + 255) & ~(size_t)255;
    return p;
  };
  float*    x     = (float*)alloc((size_t)B_ * S_ * E_ * 4);
  float*    delta = (float*)alloc((size_t)B_ * S_ * E_ * 4);
  _Float16* qh    = (_Float16*)alloc((size_t)B_ * S_ * E_ * 2);
  _Float16* qhT   = (_Float16*)alloc((size_t)B_ * E_ * S_ * 2);
  _Float16* W2T   = (_Float16*)alloc((size_t)L_ * E_ * FF_ * 2);

  k_embed<<<B_ * S_, 256, 0, stream>>>(tokens, emb, x);
  k_w2t<<<(L_ * E_ * FF_) / 256, 256, 0, stream>>>(W2, W2T);
  for (int l = 0; l < L_; l++) {
    k_q<<<(B_ * S_ * NH_) / 256, 256, 0, stream>>>(x, phi, l, qh);
    dim3 tg(S_ / 64, E_ / 64, B_);
    k_qt<<<tg, 256, 0, stream>>>(qh, qhT);
    dim3 ag(S_ / 16, B_);
    k_attn<<<ag, 256, 0, stream>>>(qh, qhT, delta);
    k_ln<<<B_ * S_, 256, 0, stream>>>(x, delta, g1 + l * E_, be1 + l * E_, x);
    k_ffn<<<(B_ * S_) / 16, 256, 0, stream>>>(x, theta, W1, b1, W2T, b2, l, delta);
    k_ln<<<B_ * S_, 256, 0, stream>>>(x, delta, g2 + l * E_, be2 + l * E_, x);
  }
  k_cls<<<B_, 256, 0, stream>>>(x, Wc, bc, out);
}
